// NanoTransformerBlock_32530082300331
// MI455X (gfx1250) — compile-verified, matrix operand order checked by reading, not run on hardware
//
#include <hip/hip_runtime.h>

// ---------------------------------------------------------------------------
// NanoTransformerBlock for MI455X (gfx1250, wave32, WMMA bf16 16x16x32)
// B=4, N=2048, D=1024, H=4096
// ---------------------------------------------------------------------------

typedef __bf16 bf16;
typedef __attribute__((ext_vector_type(16))) __bf16 v16bf;
typedef __attribute__((ext_vector_type(8)))  __bf16 v8bf;
typedef __attribute__((ext_vector_type(8)))  float  v8f;

union V16U { v16bf v16; v8bf v8[2]; };

#define BQ 4
#define NSEQ 2048
#define DMODEL 1024
#define HDIM 4096

// A/B fragment loader following ISA 16-bit 16x32 layout:
// lane L (row/col = L&15): halves 0..7 = K[kbase..kbase+7], halves 8..15 = K[16+kbase..],
// kbase = (L>=16) ? 8 : 0  ->  pass hi = ((lane>>4)<<3).
__device__ __forceinline__ v16bf load_frag(const bf16* p, int hi) {
  V16U u;
  u.v8[0] = *(const v8bf*)(p + hi);
  u.v8[1] = *(const v8bf*)(p + 16 + hi);
  return u.v16;
}

// B fragment from an LDS K x N tile (column read, lane = column).
__device__ __forceinline__ v16bf load_bcol(const bf16* p, int ld, int hi) {
  v16bf r;
#pragma unroll
  for (int j = 0; j < 8; ++j) {
    r[j]     = p[(hi + j) * ld];
    r[j + 8] = p[(16 + hi + j) * ld];
  }
  return r;
}

__device__ __forceinline__ v8f wmma_bf16(v16bf a, v16bf b, v8f c) {
  return __builtin_amdgcn_wmma_f32_16x16x32_bf16(false, a, false, b, (short)0, c,
                                                 false, false);
}

// Async global->LDS copy of 16 bytes (ASYNCcnt path) with sync fallback.
typedef int v4i_t __attribute__((vector_size(16)));
typedef __attribute__((address_space(1))) v4i_t* gv4p;
typedef __attribute__((address_space(3))) v4i_t* lv4p;

__device__ __forceinline__ void async_copy16(const bf16* g, bf16* l) {
#if __has_builtin(__builtin_amdgcn_global_load_async_to_lds_b128)
  __builtin_amdgcn_global_load_async_to_lds_b128((gv4p)g, (lv4p)l, 0, 0);
#else
  *(v8bf*)l = *(const v8bf*)g;
#endif
}

__device__ __forceinline__ void async_wait0() {
#if __has_builtin(__builtin_amdgcn_global_load_async_to_lds_b128)
#if __has_builtin(__builtin_amdgcn_s_wait_asynccnt)
  __builtin_amdgcn_s_wait_asynccnt(0);
#else
  asm volatile("s_wait_asynccnt 0x0" ::: "memory");
#endif
#endif
}

// ---------------------------------------------------------------------------
// f32 -> bf16 convert (weights)
// ---------------------------------------------------------------------------
__global__ void convert_kernel(const float* __restrict__ in, bf16* __restrict__ out,
                               long n) {
  long i = (long)blockIdx.x * blockDim.x + threadIdx.x;
  long stride = (long)gridDim.x * blockDim.x;
  for (; i < n; i += stride) out[i] = (bf16)in[i];
}

// ---------------------------------------------------------------------------
// RMSNorm: one block per row (D=1024), f32 in -> bf16 out
// ---------------------------------------------------------------------------
__global__ void rmsnorm_kernel(const float* __restrict__ x, const float* __restrict__ w,
                               bf16* __restrict__ out) {
  __shared__ float red[256];
  const size_t row = blockIdx.x;
  const int tid = threadIdx.x;
  const float* xr = x + row * DMODEL;
  float acc = 0.f;
#pragma unroll
  for (int i = tid; i < DMODEL; i += 256) { float v = xr[i]; acc += v * v; }
  red[tid] = acc;
  __syncthreads();
  for (int s = 128; s > 0; s >>= 1) {
    if (tid < s) red[tid] += red[tid + s];
    __syncthreads();
  }
  const float scale = rsqrtf(red[0] * (1.0f / DMODEL) + 1.19209290e-07f);
  bf16* orow = out + row * DMODEL;
#pragma unroll
  for (int i = tid; i < DMODEL; i += 256) orow[i] = (bf16)(xr[i] * scale * w[i]);
}

// Pipelined 16x16 score tile: S = Atile(16xD in LDS) @ hrow(16xD)^T, K = D.
__device__ __forceinline__ v8f score_tile(const bf16 (*Atile)[DMODEL],
                                          const bf16* brow, int laneLo, int hi) {
  v8f c = {};
  v16bf a  = load_frag(&Atile[laneLo][0], hi);
  v16bf bb = load_frag(brow, hi);
#pragma unroll 4
  for (int k = 32; k < DMODEL; k += 32) {
    v16bf an = load_frag(&Atile[laneLo][k], hi);
    v16bf bn = load_frag(brow + k, hi);
    c = wmma_bf16(a, bb, c);
    a = an; bb = bn;
  }
  return wmma_bf16(a, bb, c);
}

// ---------------------------------------------------------------------------
// Attention pass 1: per (batch, 16-row block) compute row max / sum-exp
// of causal-masked scores h@h^T / 32, via WMMA + online softmax stats.
// ---------------------------------------------------------------------------
__global__ void attn_stats_kernel(const bf16* __restrict__ h,
                                  float* __restrict__ rowmax,
                                  float* __restrict__ rowsum) {
  const int b = blockIdx.x, rb = blockIdx.y;
  const int tid = threadIdx.x, wave = tid >> 5, lane = tid & 31;
  const int hi = (lane >> 4) << 3;
  __shared__ __attribute__((aligned(16))) bf16 Atile[16][DMODEL];
  __shared__ float dump[8][16][17];
  __shared__ float wm[8][16], wl[8][16];
  const bf16* hb = h + (size_t)b * NSEQ * DMODEL;
  {
    int row = tid >> 4, seg = (tid & 15) * 64;
    const bf16* src = hb + (size_t)(rb * 16 + row) * DMODEL + seg;
    bf16* dst = &Atile[row][seg];
#pragma unroll
    for (int j = 0; j < 8; ++j)
      *(v8bf*)(dst + j * 8) = *(const v8bf*)(src + j * 8);
  }
  __syncthreads();
  float m_r = -3.0e38f, l_r = 0.f;
  for (int ct = wave; ct <= rb; ct += 8) {
    const bf16* brow = hb + (size_t)(ct * 16 + (lane & 15)) * DMODEL;
    v8f c = score_tile(Atile, brow, lane & 15, hi);
#pragma unroll
    for (int j = 0; j < 8; ++j) {
      float s = c[j] * (1.0f / 32.0f);
      int rg = j + hi, cg = lane & 15;
      if (ct == rb && cg > rg) s = -1.0e9f;
      dump[wave][rg][cg] = s;
    }
    asm volatile("s_wait_dscnt 0" ::: "memory");
    if (lane < 16) {
      float tm = -3.0e38f;
#pragma unroll
      for (int n = 0; n < 16; ++n) tm = fmaxf(tm, dump[wave][lane][n]);
      float nm = fmaxf(m_r, tm);
      float acc = 0.f;
#pragma unroll
      for (int n = 0; n < 16; ++n) acc += __expf(dump[wave][lane][n] - nm);
      l_r = l_r * __expf(m_r - nm) + acc;
      m_r = nm;
    }
  }
  if (lane < 16) { wm[wave][lane] = m_r; wl[wave][lane] = l_r; }
  __syncthreads();
  if (tid < 16) {
    float gm = -3.0e38f;
#pragma unroll
    for (int w = 0; w < 8; ++w) gm = fmaxf(gm, wm[w][tid]);
    float gl = 0.f;
#pragma unroll
    for (int w = 0; w < 8; ++w) gl += wl[w][tid] * __expf(wm[w][tid] - gm);
    rowmax[(size_t)b * NSEQ + rb * 16 + tid] = gm;
    rowsum[(size_t)b * NSEQ + rb * 16 + tid] = gl;
  }
}

// ---------------------------------------------------------------------------
// Attention pass 2: recompute scores, write P = exp(s-m)/l (bf16, full NxN,
// zeros above the diagonal) so the P@h GEMM can run with a plain K loop.
// ---------------------------------------------------------------------------
__global__ void attn_prob_kernel(const bf16* __restrict__ h,
                                 const float* __restrict__ rowmax,
                                 const float* __restrict__ rowsum,
                                 bf16* __restrict__ P) {
  const int b = blockIdx.x, rb = blockIdx.y;
  const int tid = threadIdx.x, wave = tid >> 5, lane = tid & 31;
  const int hi = (lane >> 4) << 3;
  __shared__ __attribute__((aligned(16))) bf16 Atile[16][DMODEL];
  const bf16* hb = h + (size_t)b * NSEQ * DMODEL;
  {
    int row = tid >> 4, seg = (tid & 15) * 64;
    const bf16* src = hb + (size_t)(rb * 16 + row) * DMODEL + seg;
    bf16* dst = &Atile[row][seg];
#pragma unroll
    for (int j = 0; j < 8; ++j)
      *(v8bf*)(dst + j * 8) = *(const v8bf*)(src + j * 8);
  }
  __syncthreads();
  float mj[8], ilj[8];
#pragma unroll
  for (int j = 0; j < 8; ++j) {
    int rg = rb * 16 + j + hi;
    mj[j] = rowmax[(size_t)b * NSEQ + rg];
    ilj[j] = 1.0f / rowsum[(size_t)b * NSEQ + rg];
  }
  bf16* Pb = P + (size_t)b * NSEQ * NSEQ;
  for (int ct = wave; ct < NSEQ / 16; ct += 8) {
    const int cg = ct * 16 + (lane & 15);
    if (ct > rb) {
#pragma unroll
      for (int j = 0; j < 8; ++j)
        Pb[(size_t)(rb * 16 + j + hi) * NSEQ + cg] = (bf16)0.0f;
    } else {
      const bf16* brow = hb + (size_t)cg * DMODEL;
      v8f c = score_tile(Atile, brow, lane & 15, hi);
#pragma unroll
      for (int j = 0; j < 8; ++j) {
        float s = c[j] * (1.0f / 32.0f);
        int rg = j + hi;
        if (ct == rb && (lane & 15) > rg) s = -1.0e9f;
        float p = __expf(s - mj[j]) * ilj[j];
        Pb[(size_t)(rb * 16 + rg) * NSEQ + cg] = (bf16)p;
      }
    }
  }
}

// ---------------------------------------------------------------------------
// Generic GEMM, B stored K x N row-major ("NN"): C[M,N] = A[M,K] @ B[K,N].
// Double-buffered LDS tiles, async-to-LDS staging overlapped with WMMAs.
// Each wave computes 16x32 (2 tiles). mode 0 -> f32 out, mode 1 -> bf16 out.
// grid: (N/256, M/16, Z), 256 threads (8 waves).
// ---------------------------------------------------------------------------
__global__ void gemm_nn_kernel(const bf16* __restrict__ A, long aZ, int lda,
                               const bf16* __restrict__ Bm, long bZ, int ldb,
                               float* __restrict__ Of, bf16* __restrict__ Ob,
                               long oZ, int ldo, int K, int mode) {
  __shared__ __attribute__((aligned(16))) bf16 Bt[2][32][256];
  __shared__ __attribute__((aligned(16))) bf16 At[2][16][32];
  const int z = blockIdx.z;
  A += (size_t)z * aZ; Bm += (size_t)z * bZ;
  const int tid = threadIdx.x, wave = tid >> 5, lane = tid & 31;
  const int hi = (lane >> 4) << 3;
  const int row0 = blockIdx.y * 16, col0 = blockIdx.x * 256;
  const int br = tid >> 3, bs = (tid & 7) * 32;           // B-tile stage mapping
  const int ar = tid >> 2, as = (tid & 3) * 8;            // A-tile stage mapping

  auto stage = [&](int buf, int k) {
    const bf16* src = Bm + (size_t)(k + br) * ldb + col0 + bs;
    bf16* dst = &Bt[buf][br][bs];
    async_copy16(src, dst);
    async_copy16(src + 8, dst + 8);
    async_copy16(src + 16, dst + 16);
    async_copy16(src + 24, dst + 24);
    if (tid < 64)
      *(v8bf*)&At[buf][ar][as] = *(const v8bf*)(A + (size_t)(row0 + ar) * lda + k + as);
  };

  v8f c0 = {}, c1 = {};
  stage(0, 0);
  async_wait0();
  __syncthreads();
  int cur = 0;
  for (int k = 32; k < K; k += 32) {
    stage(cur ^ 1, k);  // async DMA of next tile overlaps current WMMAs
    v16bf a  = load_frag(&At[cur][lane & 15][0], hi);
    v16bf b0 = load_bcol(&Bt[cur][0][wave * 32 + (lane & 15)], 256, hi);
    v16bf b1 = load_bcol(&Bt[cur][0][wave * 32 + 16 + (lane & 15)], 256, hi);
    c0 = wmma_bf16(a, b0, c0);
    c1 = wmma_bf16(a, b1, c1);
    async_wait0();
    __syncthreads();
    cur ^= 1;
  }
  {
    v16bf a  = load_frag(&At[cur][lane & 15][0], hi);
    v16bf b0 = load_bcol(&Bt[cur][0][wave * 32 + (lane & 15)], 256, hi);
    v16bf b1 = load_bcol(&Bt[cur][0][wave * 32 + 16 + (lane & 15)], 256, hi);
    c0 = wmma_bf16(a, b0, c0);
    c1 = wmma_bf16(a, b1, c1);
  }
#pragma unroll
  for (int t = 0; t < 2; ++t) {
    const int col = col0 + wave * 32 + t * 16 + (lane & 15);
    v8f cc = t ? c1 : c0;
#pragma unroll
    for (int j = 0; j < 8; ++j) {
      const size_t row = row0 + j + hi;
      if (mode == 0) Of[(size_t)z * oZ + row * ldo + col] = cc[j];
      else           Ob[(size_t)z * oZ + row * ldo + col] = (bf16)cc[j];
    }
  }
}

// ---------------------------------------------------------------------------
// GEMM against row-major weights ("NT"): C[M,N] = A[M,K] @ W[N,K]^T.
// Each wave computes 16x32 (2 N-tiles); K loop software-pipelined.
// mode 0: bf16 store | mode 1: swiglu(silu(c)*c3) dual-accum | mode 2:
// bf16 store remapped to time-major (n, b, d) for the scan tree.
// grid: (N/256, M/16), 256 threads.
// ---------------------------------------------------------------------------
__global__ void gemm_nt_kernel(const bf16* __restrict__ A, int lda,
                               const bf16* __restrict__ W1,
                               const bf16* __restrict__ W3, int ldw,
                               bf16* __restrict__ O, int ldo, int K, int mode) {
  const int tid = threadIdx.x, wave = tid >> 5, lane = tid & 31;
  const int hi = (lane >> 4) << 3;
  const int row0 = blockIdx.y * 16;
  const int colb = blockIdx.x * 256 + wave * 32;
  const bf16* arow = A + (size_t)(row0 + (lane & 15)) * lda;
  const bf16* wr0 = W1 + (size_t)(colb + (lane & 15)) * ldw;
  const bf16* wr1 = W1 + (size_t)(colb + 16 + (lane & 15)) * ldw;
  const bf16* xr0 = (mode == 1) ? (W3 + (size_t)(colb + (lane & 15)) * ldw) : wr0;
  const bf16* xr1 = (mode == 1) ? (W3 + (size_t)(colb + 16 + (lane & 15)) * ldw) : wr1;
  v8f c0 = {}, c1 = {}, c2 = {}, c3 = {};
  v16bf a  = load_frag(arow, hi);
  v16bf w0 = load_frag(wr0, hi);
  v16bf w1f = load_frag(wr1, hi);
  v16bf x0 = {}, x1 = {};
  if (mode == 1) { x0 = load_frag(xr0, hi); x1 = load_frag(xr1, hi); }
  for (int k = 32; k < K; k += 32) {
    v16bf an  = load_frag(arow + k, hi);
    v16bf w0n = load_frag(wr0 + k, hi);
    v16bf w1n = load_frag(wr1 + k, hi);
    v16bf x0n = {}, x1n = {};
    if (mode == 1) { x0n = load_frag(xr0 + k, hi); x1n = load_frag(xr1 + k, hi); }
    __builtin_prefetch(wr0 + k + 480, 0, 0);
    c0 = wmma_bf16(a, w0, c0);
    c1 = wmma_bf16(a, w1f, c1);
    if (mode == 1) { c2 = wmma_bf16(a, x0, c2); c3 = wmma_bf16(a, x1, c3); }
    a = an; w0 = w0n; w1f = w1n; x0 = x0n; x1 = x1n;
  }
  c0 = wmma_bf16(a, w0, c0);
  c1 = wmma_bf16(a, w1f, c1);
  if (mode == 1) { c2 = wmma_bf16(a, x0, c2); c3 = wmma_bf16(a, x1, c3); }
#pragma unroll
  for (int t = 0; t < 2; ++t) {
    const int col = colb + t * 16 + (lane & 15);
    v8f cg = t ? c1 : c0;
    v8f cs = t ? c3 : c2;
#pragma unroll
    for (int j = 0; j < 8; ++j) {
      const size_t row = row0 + j + hi;
      if (mode == 1) {
        float xv = cg[j];
        O[row * ldo + col] = (bf16)((xv / (1.0f + __expf(-xv))) * cs[j]);
      } else if (mode == 2) {
        size_t b = row >> 11, n = row & (NSEQ - 1);
        O[(n * BQ + b) * DMODEL + col] = (bf16)cg[j];
      } else {
        O[row * ldo + col] = (bf16)cg[j];
      }
    }
  }
}

// ---------------------------------------------------------------------------
// Scan tree combine, level l: out_p = left_p @ Q^T + right_p, where elements
// are (B=4 x D) matrices stored contiguously (elem stride 4096). 4 pairs are
// packed into one 16-row WMMA tile (row m: pair m>>2, batch m&3).
// grid: (ceil(pairs/4), D/128), 256 threads.
// ---------------------------------------------------------------------------
__global__ void scan_combine_kernel(const bf16* __restrict__ bufIn,
                                    const bf16* __restrict__ Q,
                                    bf16* __restrict__ bufOut, int numPairs) {
  const int tid = threadIdx.x, wave = tid >> 5, lane = tid & 31;
  const int hi = (lane >> 4) << 3;
  const int col = blockIdx.y * 128 + wave * 16 + (lane & 15);
  const int myq = (lane & 15) >> 2, mybatch = lane & 3;
  int p = blockIdx.x * 4 + myq;
  int pc = p < numPairs ? p : numPairs - 1;
  const bf16* leftRow = bufIn + (size_t)(2 * pc) * (BQ * DMODEL) + (size_t)mybatch * DMODEL;
  const bf16* qrow = Q + (size_t)col * DMODEL;
  v8f c;
#pragma unroll
  for (int j = 0; j < 8; ++j) {  // init C with right element
    int m = j + hi, qj = m >> 2, bj = m & 3;
    int pj = blockIdx.x * 4 + qj;
    int pcj = pj < numPairs ? pj : numPairs - 1;
    c[j] = (float)bufIn[(size_t)(2 * pcj + 1) * (BQ * DMODEL) + (size_t)bj * DMODEL + col];
  }
  v16bf a = load_frag(leftRow, hi);
  v16bf q = load_frag(qrow, hi);
  for (int k = 32; k < DMODEL; k += 32) {
    v16bf an = load_frag(leftRow + k, hi);
    v16bf qn = load_frag(qrow + k, hi);
    c = wmma_bf16(a, q, c);
    a = an; q = qn;
  }
  c = wmma_bf16(a, q, c);
#pragma unroll
  for (int j = 0; j < 8; ++j) {
    int m = j + hi, qj = m >> 2, bj = m & 3;
    int pj = blockIdx.x * 4 + qj;
    if (pj < numPairs)
      bufOut[(size_t)pj * (BQ * DMODEL) + (size_t)bj * DMODEL + col] = (bf16)c[j];
  }
}

// ---------------------------------------------------------------------------
// out[b,n,:] = x[b,n+1,:] for n<N-1 ; out[b,N-1,:] = merged[b,:]
// ---------------------------------------------------------------------------
__global__ void final_out_kernel(const float* __restrict__ x,
                                 const bf16* __restrict__ merged,
                                 float* __restrict__ out) {
  size_t idx = (size_t)blockIdx.x * blockDim.x + threadIdx.x;
  size_t b = idx / ((size_t)NSEQ * DMODEL);
  size_t r = idx % ((size_t)NSEQ * DMODEL);
  size_t n = r / DMODEL, d = r % DMODEL;
  float v;
  if (n < NSEQ - 1) v = x[b * NSEQ * DMODEL + (n + 1) * DMODEL + d];
  else              v = (float)merged[b * DMODEL + d];
  out[idx] = v;
}

// ---------------------------------------------------------------------------
// host-side orchestration
// ---------------------------------------------------------------------------
extern "C" void kernel_launch(void* const* d_in, const int* in_sizes, int n_in,
                              void* d_out, int out_size, void* d_ws, size_t ws_size,
                              hipStream_t stream) {
  const float* x     = (const float*)d_in[0];
  const float* rms_w = (const float*)d_in[1];
  // d_in[2] (ffn_rms_w) is unused by the reference graph
  const float* w1 = (const float*)d_in[3];
  const float* w3 = (const float*)d_in[4];
  const float* w2 = (const float*)d_in[5];
  const float* m0 = (const float*)d_in[6];
  const float* m1 = (const float*)d_in[7];
  float* out = (float*)d_out;
  char* ws = (char*)d_ws;

  size_t off = 0;
  auto alloc = [&](size_t bytes) { void* p = ws + off; off += (bytes + 255) & ~255ull; return p; };
  bf16*  h_bf   = (bf16*)alloc((size_t)BQ * NSEQ * DMODEL * 2);
  bf16*  g_bf   = (bf16*)alloc((size_t)BQ * NSEQ * DMODEL * 2);
  bf16*  P      = (bf16*)alloc((size_t)BQ * NSEQ * NSEQ * 2);
  float* attn_f = (float*)alloc((size_t)BQ * NSEQ * DMODEL * 4);
  bf16*  u      = (bf16*)alloc((size_t)BQ * NSEQ * HDIM * 2);
  bf16*  ffn    = (bf16*)alloc((size_t)BQ * NSEQ * DMODEL * 2);
  bf16*  w1b    = (bf16*)alloc((size_t)HDIM * DMODEL * 2);
  bf16*  w3b    = (bf16*)alloc((size_t)HDIM * DMODEL * 2);
  bf16*  w2b    = (bf16*)alloc((size_t)DMODEL * HDIM * 2);
  bf16*  m1b    = (bf16*)alloc((size_t)DMODEL * DMODEL * 2);
  bf16*  qA     = (bf16*)alloc((size_t)DMODEL * DMODEL * 2);
  bf16*  qB     = (bf16*)alloc((size_t)DMODEL * DMODEL * 2);
  bf16*  treeA  = (bf16*)alloc((size_t)NSEQ * BQ * DMODEL * 2);
  bf16*  treeB  = (bf16*)alloc((size_t)(NSEQ / 2) * BQ * DMODEL * 2);
  float* rowmax = (float*)alloc((size_t)BQ * NSEQ * 4);
  float* rowsum = (float*)alloc((size_t)BQ * NSEQ * 4);

  // weight conversions (Q_0 = m0 in "apply = @Q^T" convention)
  convert_kernel<<<1024, 256, 0, stream>>>(w1, w1b, (long)HDIM * DMODEL);
  convert_kernel<<<1024, 256, 0, stream>>>(w3, w3b, (long)HDIM * DMODEL);
  convert_kernel<<<1024, 256, 0, stream>>>(w2, w2b, (long)DMODEL * HDIM);
  convert_kernel<<<512, 256, 0, stream>>>(m1, m1b, (long)DMODEL * DMODEL);
  convert_kernel<<<512, 256, 0, stream>>>(m0, qA, (long)DMODEL * DMODEL);

  // h = rmsnorm(x)
  rmsnorm_kernel<<<BQ * NSEQ, 256, 0, stream>>>(x, rms_w, h_bf);

  // attention
  attn_stats_kernel<<<dim3(BQ, NSEQ / 16), 256, 0, stream>>>(h_bf, rowmax, rowsum);
  attn_prob_kernel<<<dim3(BQ, NSEQ / 16), 256, 0, stream>>>(h_bf, rowmax, rowsum, P);
  gemm_nn_kernel<<<dim3(DMODEL / 256, NSEQ / 16, BQ), 256, 0, stream>>>(
      P, (long)NSEQ * NSEQ, NSEQ, h_bf, (long)NSEQ * DMODEL, DMODEL,
      attn_f, nullptr, (long)NSEQ * DMODEL, DMODEL, NSEQ, 0);

  // g = rmsnorm(attn_out)
  rmsnorm_kernel<<<BQ * NSEQ, 256, 0, stream>>>(attn_f, rms_w, g_bf);

  // FFN: u = silu(g@w1^T) * (g@w3^T) ; ffn = u@w2^T
  gemm_nt_kernel<<<dim3(HDIM / 256, BQ * NSEQ / 16), 256, 0, stream>>>(
      g_bf, DMODEL, w1b, w3b, DMODEL, u, HDIM, DMODEL, 1);
  gemm_nt_kernel<<<dim3(DMODEL / 256, BQ * NSEQ / 16), 256, 0, stream>>>(
      u, HDIM, w2b, nullptr, HDIM, ffn, DMODEL, HDIM, 0);

  // y_i = ffn_i @ m1^T, stored time-major as scan-tree leaves
  gemm_nt_kernel<<<dim3(DMODEL / 256, BQ * NSEQ / 16), 256, 0, stream>>>(
      ffn, DMODEL, m1b, nullptr, DMODEL, treeA, DMODEL, DMODEL, 2);

  // 11-level binary-tree fold: b = b_left @ Q_l^T + b_right, Q_{l+1} = Q_l @ Q_l
  bf16 *cur = treeA, *nxt = treeB, *qcur = qA, *qnxt = qB;
  for (int l = 0; l < 11; ++l) {
    int pairs = NSEQ >> (l + 1);
    scan_combine_kernel<<<dim3((pairs + 3) / 4, DMODEL / 128), 256, 0, stream>>>(
        cur, qcur, nxt, pairs);
    if (l < 10) {
      gemm_nn_kernel<<<dim3(DMODEL / 256, DMODEL / 16, 1), 256, 0, stream>>>(
          qcur, 0, DMODEL, qcur, 0, DMODEL, nullptr, qnxt, 0, DMODEL, DMODEL, 1);
      bf16* t = qcur; qcur = qnxt; qnxt = t;
    }
    bf16* t = cur; cur = nxt; nxt = t;
  }

  // out = concat(x[:,1:,:], merged)
  final_out_kernel<<<(BQ * NSEQ * DMODEL) / 256, 256, 0, stream>>>(x, cur, out);
}
